// Transform_7705171329026
// MI455X (gfx1250) — matrix operand order repaired and checked by reading, not compiled, not run on hardware
//
#include <hip/hip_runtime.h>
#include <hip/hip_bf16.h>

typedef __attribute__((ext_vector_type(16))) __bf16 v16bf;
typedef __attribute__((ext_vector_type(8)))  __bf16 v8bf;
typedef __attribute__((ext_vector_type(4)))  __bf16 v4bf;
typedef __attribute__((ext_vector_type(8)))  float  v8f;
typedef __attribute__((ext_vector_type(4)))  unsigned int u32x4;
typedef __attribute__((ext_vector_type(8)))  int i32x8;
typedef __attribute__((ext_vector_type(4)))  int i32x4;

#define B_  8
#define T_  48
#define HZ  24
#define NN  1024
#define HH  512
#define TC  64
#define NH  8
#define DD  64
#define NT  2   // n-columns per workgroup

#if defined(__has_builtin)
#if __has_builtin(__builtin_amdgcn_tensor_load_to_lds) && \
    __has_builtin(__builtin_amdgcn_s_wait_tensorcnt)
#define HAVE_TDM 1
#endif
#endif

// ---------------------------------------------------------------------------
// Kernel 0: attention weights  attn[b,h,i,j] = softmax_j( q.k / 8 )
// One block of 64 threads per (b,h).  Tiny (~26 MFLOP total).
// ---------------------------------------------------------------------------
__global__ __launch_bounds__(64) void attn_kernel(
    const float* __restrict__ x_time, const float* __restrict__ target_time,
    const float* __restrict__ Wq, const float* __restrict__ bq,
    const float* __restrict__ Wk, const float* __restrict__ bk,
    float* __restrict__ attn)
{
  const int b = blockIdx.x, h = blockIdx.y;
  __shared__ float kh[T_][DD];
  __shared__ float qh[HZ][DD];
  const int tid = threadIdx.x;
  for (int idx = tid; idx < T_ * DD; idx += 64) {
    const int j = idx >> 6, dd = idx & 63;
    float acc = bk[h * DD + dd];
    const float* xr = x_time + (b * T_ + j) * TC;
    #pragma unroll 8
    for (int c = 0; c < TC; ++c) acc += xr[c] * Wk[c * HH + h * DD + dd];
    kh[j][dd] = acc;
  }
  for (int idx = tid; idx < HZ * DD; idx += 64) {
    const int i = idx >> 6, dd = idx & 63;
    float acc = bq[h * DD + dd];
    const float* tr = target_time + (b * HZ + i) * TC;
    #pragma unroll 8
    for (int c = 0; c < TC; ++c) acc += tr[c] * Wq[c * HH + h * DD + dd];
    qh[i][dd] = acc;
  }
  __syncthreads();
  if (tid < HZ) {
    const int i = tid;
    float s[T_];
    float mx = -3.4e38f;
    #pragma unroll 4
    for (int j = 0; j < T_; ++j) {
      float acc = 0.f;
      #pragma unroll 8
      for (int dd = 0; dd < DD; ++dd) acc += qh[i][dd] * kh[j][dd];
      acc *= 0.125f;             // 1/sqrt(d), d = 64
      s[j] = acc;
      mx = fmaxf(mx, acc);
    }
    float sum = 0.f;
    #pragma unroll 4
    for (int j = 0; j < T_; ++j) { s[j] = __expf(s[j] - mx); sum += s[j]; }
    const float inv = 1.f / sum;
    float* orow = attn + ((b * NH + h) * HZ + i) * T_;
    #pragma unroll 4
    for (int j = 0; j < T_; ++j) orow[j] = s[j] * inv;
  }
}

// ---------------------------------------------------------------------------
// Kernel 1: Wvt[c'][c] = bf16(Wv[c][c'])   (transpose + convert, 512x512)
// ---------------------------------------------------------------------------
__global__ __launch_bounds__(256) void wvt_kernel(
    const float* __restrict__ Wv, __bf16* __restrict__ Wvt)
{
  const int idx = blockIdx.x * 256 + threadIdx.x;
  const int cp = idx >> 9;    // output row  (Wv column)
  const int c  = idx & 511;   // output col  (Wv row)
  Wvt[cp * HH + c] = (__bf16)Wv[c * HH + cp];
}

// ---------------------------------------------------------------------------
// Main fused kernel helpers
// ---------------------------------------------------------------------------
__device__ inline v16bf lds_afrag(const __bf16* p) {
  // A-fragment per lane: K = base+[0..7] and base+[16..23]  (ISA 7.12.2)
  v8bf lo = *(const v8bf*)p;
  v8bf hi = *(const v8bf*)(p + 16);
  return __builtin_shufflevector(lo, hi, 0,1,2,3,4,5,6,7,8,9,10,11,12,13,14,15);
}
__device__ inline v16bf lds_bfrag(const __bf16* p) {
  // B-fragment per lane: K = base+[0..7] and base+[16..23] of one column (stored transposed)
  v8bf lo = *(const v8bf*)p;
  v8bf hi = *(const v8bf*)(p + 16);
  return __builtin_shufflevector(lo, hi, 0,1,2,3,4,5,6,7,8,9,10,11,12,13,14,15);
}
__device__ inline v8f wmma_bf16(v16bf a, v16bf b, v8f c) {
  return __builtin_amdgcn_wmma_f32_16x16x32_bf16(false, a, false, b, (short)0, c,
                                                 false, false);
}

// ---------------------------------------------------------------------------
// Kernel 2: fused  V = enc@Wv+bv ; out = attn@V   per (b, n-pair)
// Block = 256 threads (8 wave32).  Grid = (N/NT, B) = (512, 8).
// ---------------------------------------------------------------------------
__global__ __launch_bounds__(256) void fused_attn_v(
    const float*  __restrict__ enc,     // [B,T,N,H] fp32
    const float*  __restrict__ attnG,   // [B,nh,Hz,T] fp32 (ws)
    const __bf16* __restrict__ WvtG,    // [H,H] bf16 transposed (ws)
    const float*  __restrict__ bv,      // [H]
    float*        __restrict__ out)     // [B,Hz,N,H] fp32
{
  __shared__ __bf16 As[NT * T_ * HH];   // 96 x 512  A (bf16)             96 KB
  __shared__ __bf16 Wt[DD * HH];        // 64 x 512  Wv chunk, transposed 64 KB
  __shared__ __bf16 Attn_s[32 * 64];    // padded attn, current head       4 KB
  __shared__ __bf16 Vs[NT * DD * 64];   // [n][c'][j_pad] V transposed    16 KB

  const int b    = blockIdx.y;
  const int ng   = blockIdx.x;
  const int tid  = threadIdx.x;
  const int wave = tid >> 5;
  const int lane = tid & 31;
  const int lrow = lane & 15;
  const int lhi  = lane >> 4;

  // ---- phase 0: zero V pad, stage A slice (fp32 -> bf16) ----
  for (int idx = tid; idx < NT * DD * 64; idx += 256) Vs[idx] = (__bf16)0.f;
  for (int idx = tid; idx < NT * T_ * (HH / 4); idx += 256) {
    const int row = idx >> 7;           // / 128
    const int c4  = idx & 127;
    const int nl  = row / T_, t = row % T_;
    const int n   = ng * NT + nl;
    const float4 f =
        *(const float4*)(enc + (((size_t)(b * T_ + t) * NN + n) * HH + c4 * 4));
    v4bf p;
    p[0] = (__bf16)f.x; p[1] = (__bf16)f.y; p[2] = (__bf16)f.z; p[3] = (__bf16)f.w;
    *(v4bf*)&As[row * HH + c4 * 4] = p;
  }
  __syncthreads();

  for (int h = 0; h < NH; ++h) {
    if (h) __syncthreads();             // protect Wt/Attn_s/Vs reuse

    // ---- load transposed Wv column-block for this head (64 KB, contiguous) ----
#ifdef HAVE_TDM
    if (wave == 0) {
      // Tensor Data Mover: 1-D tile of 32768 bf16 elements, L2 -> LDS.
      const unsigned lds_off = (unsigned)(uintptr_t)(&Wt[0]);
      const unsigned long long ga =
          (unsigned long long)(uintptr_t)(WvtG + (size_t)h * DD * HH);
      u32x4 g0;
      g0[0] = 1u;                                   // count=1, user descriptor
      g0[1] = lds_off;                              // lds_addr (bytes)
      g0[2] = (unsigned)(ga & 0xFFFFFFFFu);         // global_addr[31:0]
      g0[3] = (unsigned)((ga >> 32) & 0x01FFFFFFu)  // global_addr[56:32]
              | (2u << 30);                         // type = 2 ("image")
      i32x8 g1;
      g1[0] = 0x00010000;          // workgroup_mask=0, data_size=1 (2 bytes)
      g1[1] = (int)(0x8000u << 16);// tensor_dim0 = 32768 (lo16 @ [31:16])
      g1[2] = (int)(1u << 16);     // tensor_dim0 hi = 0, tensor_dim1 = 1
      g1[3] = (int)(0x8000u << 16);// tensor_dim1 hi = 0, tile_dim0 = 32768
      g1[4] = 0;                   // tile_dim1 = 0 (unused), tile_dim2 = 0
      g1[5] = 0x8000;              // tensor_dim0_stride = 32768 (lo 32)
      g1[6] = 0;                   // stride hi, tensor_dim1_stride lo
      g1[7] = 0;
      const i32x4 z4 = {0, 0, 0, 0};
#if defined(__clang_major__) && __clang_major__ >= 23
      const i32x8 z8 = {0, 0, 0, 0, 0, 0, 0, 0};
      __builtin_amdgcn_tensor_load_to_lds(g0, g1, z4, z4, z8, 0);
#else
      __builtin_amdgcn_tensor_load_to_lds(g0, g1, z4, z4, 0);
#endif
      __builtin_amdgcn_s_wait_tensorcnt(0);   // wave 0 waits; barrier releases rest
    }
#else
    for (int idx = tid; idx < DD * (HH / 8); idx += 256) {
      const int r  = idx >> 6;          // / 64
      const int ch = idx & 63;
      *(v8bf*)&Wt[r * HH + ch * 8] =
          *(const v8bf*)(WvtG + (size_t)(h * DD + r) * HH + ch * 8);
    }
#endif
    // load padded attn [32 x 64] (i<24, j<48 valid, rest zero)
    for (int idx = tid; idx < 32 * 64; idx += 256) {
      const int i = idx >> 6, j = idx & 63;
      float a = 0.f;
      if (i < HZ && j < T_) a = attnG[((b * NH + h) * HZ + i) * T_ + j];
      Attn_s[idx] = (__bf16)a;
    }
    __syncthreads();

    // ---- stage 1: V[96 x 64] = A[96 x 512] @ Wv_h[512 x 64]  ----
    // wave -> ntile = wave&3, mtiles {m0, m0+2, m0+4}, m0 = wave>>2
    const int nt = wave & 3;
    const int m0 = wave >> 2;
    v8f acc0 = {}, acc1 = {}, acc2 = {};
    const __bf16* Brow = &Wt[(nt * 16 + lrow) * HH];
    #pragma unroll 4
    for (int ks = 0; ks < HH / 32; ++ks) {
      const int kb = ks * 32 + lhi * 8;
      const v16bf bf = lds_bfrag(Brow + kb);
      const int ka = ks * 32 + lhi * 8;
      const v16bf a0 = lds_afrag(&As[((m0    ) * 16 + lrow) * HH + ka]);
      const v16bf a1 = lds_afrag(&As[((m0 + 2) * 16 + lrow) * HH + ka]);
      const v16bf a2 = lds_afrag(&As[((m0 + 4) * 16 + lrow) * HH + ka]);
      acc0 = wmma_bf16(a0, bf, acc0);
      acc1 = wmma_bf16(a1, bf, acc1);
      acc2 = wmma_bf16(a2, bf, acc2);
    }
    // stash V tiles to LDS, bf16, transposed ([c'][j]); 8 contiguous j per lane
    {
      const v8f accs[3] = {acc0, acc1, acc2};
      #pragma unroll
      for (int s = 0; s < 3; ++s) {
        const int m = m0 + 2 * s;
        const int rowbase = m * 16 + 8 * lhi;     // multiple of 8, within one n
        const int nl = rowbase / T_;
        const int jb = rowbase - nl * T_;
        v8bf vv;
        #pragma unroll
        for (int r = 0; r < 8; ++r) vv[r] = (__bf16)accs[s][r];
        *(v8bf*)&Vs[(nl * DD + nt * 16 + lrow) * 64 + jb] = vv;
      }
    }
    __syncthreads();

    // ---- stage 2: out[32 x 64] = attn_pad[32 x 64] @ V[64 x 64] per n ----
    const int nl2 = wave >> 2;
    const int nt2 = wave & 3;
    v8f o0 = {}, o1 = {};
    const __bf16* B2row = &Vs[(nl2 * DD + nt2 * 16 + lrow) * 64];
    #pragma unroll
    for (int ks = 0; ks < 2; ++ks) {
      const int kb = ks * 32 + lhi * 8;
      const v16bf bf = lds_bfrag(B2row + kb);
      const int ka = ks * 32 + lhi * 8;
      const v16bf aA0 = lds_afrag(&Attn_s[(lrow) * 64 + ka]);
      const v16bf aA1 = lds_afrag(&Attn_s[(16 + lrow) * 64 + ka]);
      o0 = wmma_bf16(aA0, bf, o0);
      o1 = wmma_bf16(aA1, bf, o1);
    }
    // ---- epilogue: bias + fp32 store ----
    const int n    = ng * NT + nl2;
    const int cloc = h * DD + nt2 * 16 + lrow;
    const float bias = bv[cloc];
    const v8f oaccs[2] = {o0, o1};
    #pragma unroll
    for (int mt = 0; mt < 2; ++mt) {
      const int ib = mt * 16 + 8 * lhi;
      if (ib < HZ) {                    // drop padded rows i >= 24
        #pragma unroll
        for (int r = 0; r < 8; ++r) {
          const int i = ib + r;
          out[((size_t)(b * HZ + i) * NN + n) * HH + cloc] = oaccs[mt][r] + bias;
        }
      }
    }
  }
}

// ---------------------------------------------------------------------------
extern "C" void kernel_launch(void* const* d_in, const int* in_sizes, int n_in,
                              void* d_out, int out_size, void* d_ws, size_t ws_size,
                              hipStream_t stream) {
  const float* enc   = (const float*)d_in[0];  // encoder_hidden [B,T,N,H]
  const float* x_t   = (const float*)d_in[1];  // x_time       [B,T,Tc]
  const float* tgt_t = (const float*)d_in[2];  // target_time  [B,Hz,Tc]
  const float* Wq    = (const float*)d_in[3];
  const float* bq    = (const float*)d_in[4];
  const float* Wk    = (const float*)d_in[5];
  const float* bk    = (const float*)d_in[6];
  const float* Wv    = (const float*)d_in[7];
  const float* bv    = (const float*)d_in[8];
  float* out = (float*)d_out;

  // workspace: attn fp32 [8,8,24,48] = 294912 B, then Wv^T bf16 [512,512] = 524288 B
  float*  attn = (float*)d_ws;
  __bf16* Wvt  = (__bf16*)((char*)d_ws + 294912);

  attn_kernel<<<dim3(B_, NH), 64, 0, stream>>>(x_t, tgt_t, Wq, bq, Wk, bk, attn);
  wvt_kernel<<<dim3((HH * HH) / 256), 256, 0, stream>>>(Wv, Wvt);
  fused_attn_v<<<dim3(NN / NT, B_), 256, 0, stream>>>(enc, attn, Wvt, bv, out);
}
